// RNNDecoder_20186346291814
// MI455X (gfx1250) — compile-verified, matrix operand order checked by reading, not run on hardware
//
#include <hip/hip_runtime.h>
#include <hip/hip_bf16.h>

// ---------------------------------------------------------------------------
// 3-layer LSTM (B=64, T=512, H=1024) + output projection, CDNA5 / gfx1250.
//  - Persistent kernel, 128 WGs x 256 threads (8 waves), one grid barrier per
//    timestep. Each WG owns 8 hidden units (32 gate columns); W_hh / W_ih
//    column slices live in LDS as bf16 for the whole layer.
//  - K-split wave groups: waves 0-3 compute x_t.W_ih^T, waves 4-7 compute
//    h_{t-1}.W_hh^T; each wave holds 2 accumulators (both 16-col N tiles) so
//    every A fragment is reused twice. Partial sums combined through LDS.
//  - Layer body is template<int L>-specialized so the history-buffer base
//    pointers stay uniform -> global_load (not flat_load) for A fragments,
//    decoupling LOADcnt from DScnt. K loops unrolled 4x to fill WMMA->VALU
//    hazard slots with independent fragment streams.
//  - bf16 WMMA (v_wmma_f32_16x16x32_bf16) with f32 accumulate; activations,
//    cell state in f32; h stored bf16 in [B,T,H] history buffers.
// ---------------------------------------------------------------------------

typedef __attribute__((ext_vector_type(16))) __bf16 v16bf;
typedef __attribute__((ext_vector_type(8)))  float  v8f;

#define NWG   128
#define TBLK  256
#define B_    64
#define T_    512
#define H_    1024
#define OUT_  121
#define KPAD  1032            // bf16 elems per LDS weight column (1024 + 8 pad)

struct FragU { unsigned int u[8]; };   // 8 dwords = 16 bf16

__device__ __forceinline__ unsigned short f2bf(float f) {
  unsigned u = __builtin_bit_cast(unsigned, f);
  u += 0x7FFFu + ((u >> 16) & 1u);     // round-to-nearest-even
  return (unsigned short)(u >> 16);
}

// 16x32 bf16 A-fragment from global: uniform base + 32-bit lane offset
// (elements). Lane mapping per ISA 7.12.2: lanes 0-15 hold M=lane, K pairs
// {0,1..6,7 | 16..23}; lanes 16-31 get the +8 K half (folded into `off`).
__device__ __forceinline__ v16bf load_a_g(const unsigned short* __restrict__ base,
                                          unsigned off) {
  FragU f;
#pragma unroll
  for (int v = 0; v < 8; ++v) {
    unsigned k = ((v & 3u) << 1) + ((unsigned)(v >> 2) << 4);
    f.u[v] = *(const unsigned int*)(base + (off + k));
  }
  return __builtin_bit_cast(v16bf, f);
}

// 32x16 bf16 B-fragment from LDS, weights stored col-major [col][kstride].
__device__ __forceinline__ v16bf load_b_lds(const unsigned short* __restrict__ w,
                                            int n0, int kstride, int k0, int lane) {
  FragU f;
  int n = lane & 15;
  int khalf = (lane >> 4) << 3;
  const unsigned short* cp = w + (n0 + n) * kstride + k0 + khalf;
#pragma unroll
  for (int v = 0; v < 8; ++v) {
    int k = ((v & 3) << 1) + ((v >> 2) << 4);
    f.u[v] = *(const unsigned int*)(cp + k);
  }
  return __builtin_bit_cast(v16bf, f);
}

__device__ __forceinline__ float sigmoidf_(float x) {
  return 1.0f / (1.0f + __expf(-x));
}

__device__ __forceinline__ void grid_barrier(unsigned int* __restrict__ cnt,
                                             unsigned& bar_target, int tid) {
  __syncthreads();
  __threadfence();
  bar_target += NWG;
  if (tid == 0) {
    __hip_atomic_fetch_add(cnt, 1u, __ATOMIC_RELEASE, __HIP_MEMORY_SCOPE_AGENT);
    while (__hip_atomic_load(cnt, __ATOMIC_ACQUIRE, __HIP_MEMORY_SCOPE_AGENT)
           < bar_target) {
      __builtin_amdgcn_s_sleep(1);
    }
  }
  __syncthreads();
}

// ---------------------------------------------------------------------------
// One LSTM layer, compile-time specialized so xprev/hcur/weight bases remain
// uniform kernel arguments (=> global_load for A fragments).
// ---------------------------------------------------------------------------
template <int L>
__device__ __forceinline__ void run_layer(
    const float* __restrict__ strokes,
    const float* __restrict__ Wih,  const float* __restrict__ Whh,
    const float* __restrict__ bih,  const float* __restrict__ bhh,
    const unsigned short* __restrict__ xprev,   // null for L==0
    unsigned short* __restrict__ hcur,
    unsigned short* Whh_lds, unsigned short* Wih_lds,
    float* gates0, float* gates1, float* cst, float* bias,
    unsigned int* __restrict__ barrier_cnt, unsigned& bar_target,
    int tid, int lane, int grp, int mt, int j0)
{
  const unsigned hist_row = (unsigned)T_ * H_;

  // ---- stage this WG's 32 gate-column weight slices into LDS (bf16) ----
  {
    int c  = tid >> 3;                 // column 0..31
    int ks = (tid & 7) * 128;          // 128-elem K strip
    int g = c >> 3, jj = c & 7;
    long wrow = (long)(g * H_ + j0 + jj);
    const float* src_hh = Whh + wrow * H_ + ks;
    unsigned* dst_hh = (unsigned*)(Whh_lds + c * KPAD + ks);
#pragma unroll 4
    for (int k = 0; k < 128; k += 4) {
      float4 q = *(const float4*)(src_hh + k);
      dst_hh[(k >> 1) + 0] = (unsigned)f2bf(q.x) | ((unsigned)f2bf(q.y) << 16);
      dst_hh[(k >> 1) + 1] = (unsigned)f2bf(q.z) | ((unsigned)f2bf(q.w) << 16);
    }
    if constexpr (L > 0) {
      const float* src_ih = Wih + wrow * H_ + ks;
      unsigned* dst_ih = (unsigned*)(Wih_lds + c * KPAD + ks);
#pragma unroll 4
      for (int k = 0; k < 128; k += 4) {
        float4 q = *(const float4*)(src_ih + k);
        dst_ih[(k >> 1) + 0] = (unsigned)f2bf(q.x) | ((unsigned)f2bf(q.y) << 16);
        dst_ih[(k >> 1) + 1] = (unsigned)f2bf(q.z) | ((unsigned)f2bf(q.w) << 16);
      }
    }
    if (tid < 32) {
      int gg = tid >> 3, j2 = tid & 7;
      long r = (long)(gg * H_ + j0 + j2);
      bias[tid] = bih[r] + bhh[r];
    }
    cst[tid] = 0.0f; cst[tid + 256] = 0.0f;    // reset cell state
  }
  __syncthreads();

  for (int t = 0; t < T_; ++t) {
    v8f acc0, acc1;                         // N tiles 0 and 1 (cols 0-15,16-31)
    const int n0    = lane & 15;
    const int mbase = mt * 16 + ((lane >> 4) << 3);
    const unsigned arow =
        (unsigned)(mt * 16 + n0) * hist_row + (unsigned)((lane >> 4) << 3);

    if (grp == 0) {
      // ---- x path: bias (+ layer-0 K=3 projection) (+ x_t . W_ih^T) ----
      float bv0 = bias[n0], bv1 = bias[16 + n0];
#pragma unroll
      for (int v = 0; v < 8; ++v) { acc0[v] = bv0; acc1[v] = bv1; }
      if constexpr (L == 0) {
        int g0 = n0 >> 3,        jj0 = n0 & 7;
        int g1 = (16 + n0) >> 3, jj1 = n0 & 7;
        long r0 = (long)(g0 * H_ + j0 + jj0) * 3;
        long r1 = (long)(g1 * H_ + j0 + jj1) * 3;
        float w00 = Wih[r0], w01 = Wih[r0 + 1], w02 = Wih[r0 + 2];
        float w10 = Wih[r1], w11 = Wih[r1 + 1], w12 = Wih[r1 + 2];
#pragma unroll
        for (int v = 0; v < 8; ++v) {
          const float* sp = strokes + ((long)(mbase + v) * T_ + t) * 3;
          float s0 = sp[0], s1 = sp[1], s2 = sp[2];
          acc0[v] += s0 * w00 + s1 * w01 + s2 * w02;
          acc1[v] += s0 * w10 + s1 * w11 + s2 * w12;
        }
      } else {
        unsigned abase = arow + (unsigned)t * H_;
        if (t + 1 < T_) __builtin_prefetch(xprev + abase + H_, 0, 3);
#pragma unroll 4
        for (int kk = 0; kk < H_; kk += 32) {
          v16bf a  = load_a_g(xprev, abase + kk);
          v16bf b0 = load_b_lds(Wih_lds, 0,  KPAD, kk, lane);
          v16bf b1 = load_b_lds(Wih_lds, 16, KPAD, kk, lane);
          acc0 = __builtin_amdgcn_wmma_f32_16x16x32_bf16(
                     false, a, false, b0, (short)0, acc0, false, false);
          acc1 = __builtin_amdgcn_wmma_f32_16x16x32_bf16(
                     false, a, false, b1, (short)0, acc1, false, false);
        }
      }
    } else {
      // ---- h path: h_{t-1} . W_hh^T (zero at t=0, h_init = 0) ----
#pragma unroll
      for (int v = 0; v < 8; ++v) { acc0[v] = 0.0f; acc1[v] = 0.0f; }
      if (t > 0) {
        unsigned abase = arow + (unsigned)(t - 1) * H_;
#pragma unroll 4
        for (int kk = 0; kk < H_; kk += 32) {
          v16bf a  = load_a_g(hcur, abase + kk);
          v16bf b0 = load_b_lds(Whh_lds, 0,  KPAD, kk, lane);
          v16bf b1 = load_b_lds(Whh_lds, 16, KPAD, kk, lane);
          acc0 = __builtin_amdgcn_wmma_f32_16x16x32_bf16(
                     false, a, false, b0, (short)0, acc0, false, false);
          acc1 = __builtin_amdgcn_wmma_f32_16x16x32_bf16(
                     false, a, false, b1, (short)0, acc1, false, false);
        }
      }
    }
    // ---- spill partial gate tiles (C layout: VGPR v -> M = v + 8*hi) ----
    {
      float* gdst = (grp == 0) ? gates0 : gates1;
#pragma unroll
      for (int v = 0; v < 8; ++v) {
        gdst[(mbase + v) * 33 + n0]      = acc0[v];
        gdst[(mbase + v) * 33 + 16 + n0] = acc1[v];
      }
    }
    __syncthreads();
    // ---- elementwise LSTM update: 64 batch x 8 hidden = 512 items ----
#pragma unroll
    for (int it = 0; it < 2; ++it) {
      int idx = tid + it * 256;
      int m = idx >> 3, jj = idx & 7;
      float iv = sigmoidf_(gates0[m * 33 + 0  + jj] + gates1[m * 33 + 0  + jj]);
      float fv = sigmoidf_(gates0[m * 33 + 8  + jj] + gates1[m * 33 + 8  + jj]);
      float gv = tanhf   (gates0[m * 33 + 16 + jj] + gates1[m * 33 + 16 + jj]);
      float ov = sigmoidf_(gates0[m * 33 + 24 + jj] + gates1[m * 33 + 24 + jj]);
      float c = fv * cst[idx] + iv * gv;
      cst[idx] = c;
      float h = ov * tanhf(c);
      hcur[(unsigned)m * hist_row + (unsigned)t * H_ + j0 + jj] = f2bf(h);
    }
    // ---- grid-wide barrier (monotonic count; all 128 WGs resident) ----
    grid_barrier(barrier_cnt, bar_target, tid);
  }
  __syncthreads();
}

// ---------------------------------------------------------------------------
// Persistent 3-layer LSTM kernel.
// ---------------------------------------------------------------------------
__global__ void __launch_bounds__(TBLK, 1)
lstm3_persistent(const float* __restrict__ strokes,
                 const float* __restrict__ Wih0, const float* __restrict__ Whh0,
                 const float* __restrict__ bih0, const float* __restrict__ bhh0,
                 const float* __restrict__ Wih1, const float* __restrict__ Whh1,
                 const float* __restrict__ bih1, const float* __restrict__ bhh1,
                 const float* __restrict__ Wih2, const float* __restrict__ Whh2,
                 const float* __restrict__ bih2, const float* __restrict__ bhh2,
                 unsigned short* __restrict__ hA, unsigned short* __restrict__ hB,
                 unsigned int* __restrict__ barrier_cnt)
{
  extern __shared__ char smem_raw[];
  unsigned short* Whh_lds = (unsigned short*)smem_raw;          // 32*KPAD bf16
  unsigned short* Wih_lds = Whh_lds + 32 * KPAD;                // 32*KPAD bf16
  float* gates0 = (float*)(Wih_lds + 32 * KPAD);                // 64 x 33 f32
  float* gates1 = gates0 + 64 * 33;                             // 64 x 33 f32
  float* cst    = gates1 + 64 * 33;                             // 512 f32
  float* bias   = cst + 512;                                    // 32 f32

  const int tid  = threadIdx.x;
  const int lane = tid & 31;
  const int wave = tid >> 5;
  const int grp  = wave >> 2;                 // 0: x/W_ih path, 1: h/W_hh path
  const int mt   = wave & 3;                  // M tile of 16 (B=64)
  const int j0   = blockIdx.x * 8;            // this WG's hidden-unit base

  unsigned bar_target = 0;

  run_layer<0>(strokes, Wih0, Whh0, bih0, bhh0, nullptr, hA,
               Whh_lds, Wih_lds, gates0, gates1, cst, bias,
               barrier_cnt, bar_target, tid, lane, grp, mt, j0);
  run_layer<1>(strokes, Wih1, Whh1, bih1, bhh1, hA, hB,
               Whh_lds, Wih_lds, gates0, gates1, cst, bias,
               barrier_cnt, bar_target, tid, lane, grp, mt, j0);
  run_layer<2>(strokes, Wih2, Whh2, bih2, bhh2, hB, hA,
               Whh_lds, Wih_lds, gates0, gates1, cst, bias,
               barrier_cnt, bar_target, tid, lane, grp, mt, j0);
}

// ---------------------------------------------------------------------------
// Output projection: out[BT,121] = h_final[BT,1024] . W_out^T + b_out.
// W_out staged in LDS (bf16, col-major, padded); each wave owns one M tile and
// all 8 N tiles (8 accumulators) -> every A fragment loaded exactly once.
// ---------------------------------------------------------------------------
__global__ void __launch_bounds__(256, 1)
out_proj(const unsigned short* __restrict__ hfin,
         const unsigned short* __restrict__ woutbf,
         const float* __restrict__ bout, float* __restrict__ out)
{
  extern __shared__ char smem_raw[];
  unsigned short* wlds = (unsigned short*)smem_raw;    // 128 * KPAD bf16
  const int tid = threadIdx.x, lane = tid & 31, wave = tid >> 5;

  // zero pad columns 121..127, then copy W_out bf16 ([O,H] row-major is
  // exactly col-major [n][k]) into padded LDS
  for (int i = tid; i < (128 - OUT_) * KPAD; i += TBLK)
    wlds[OUT_ * KPAD + i] = 0;
  for (int i = tid; i < OUT_ * (H_ / 2); i += TBLK) {
    int n = i / (H_ / 2), k2 = i - n * (H_ / 2);
    *(unsigned*)(wlds + n * KPAD + (k2 << 1)) =
        *(const unsigned*)(woutbf + n * H_ + (k2 << 1));
  }
  __syncthreads();

  const int mtile = blockIdx.x * 8 + wave;             // 0..2047 (BT/16)
  const unsigned abase =
      (unsigned)(mtile * 16 + (lane & 15)) * H_ + (unsigned)((lane >> 4) << 3);

  v8f acc[8];
#pragma unroll
  for (int nt = 0; nt < 8; ++nt) {
    int n = nt * 16 + (lane & 15);
    float bv = (n < OUT_) ? bout[n] : 0.0f;
#pragma unroll
    for (int v = 0; v < 8; ++v) acc[nt][v] = bv;
  }
#pragma unroll 2
  for (int kk = 0; kk < H_; kk += 32) {
    v16bf a = load_a_g(hfin, abase + kk);
#pragma unroll
    for (int nt = 0; nt < 8; ++nt) {
      v16bf b = load_b_lds(wlds, nt * 16, KPAD, kk, lane);
      acc[nt] = __builtin_amdgcn_wmma_f32_16x16x32_bf16(
                    false, a, false, b, (short)0, acc[nt], false, false);
    }
  }
  const int mbase = mtile * 16 + ((lane >> 4) << 3);
#pragma unroll
  for (int nt = 0; nt < 8; ++nt) {
    int n = nt * 16 + (lane & 15);
    if (n < OUT_) {
#pragma unroll
      for (int v = 0; v < 8; ++v) out[(long)(mbase + v) * OUT_ + n] = acc[nt][v];
    }
  }
}

// ---------------------------------------------------------------------------
__global__ void cvt_f32_bf16(const float* __restrict__ src,
                             unsigned short* __restrict__ dst, int n) {
  int i = blockIdx.x * blockDim.x + threadIdx.x;
  if (i < n) dst[i] = f2bf(src[i]);
}

__global__ void zero_u32(unsigned int* p) {
  if (threadIdx.x == 0 && blockIdx.x == 0) *p = 0u;
}

// ---------------------------------------------------------------------------
extern "C" void kernel_launch(void* const* d_in, const int* in_sizes, int n_in,
                              void* d_out, int out_size, void* d_ws, size_t ws_size,
                              hipStream_t stream) {
  const float* strokes = (const float*)d_in[0];
  const float* Wih0 = (const float*)d_in[1];
  const float* Whh0 = (const float*)d_in[2];
  const float* bih0 = (const float*)d_in[3];
  const float* bhh0 = (const float*)d_in[4];
  const float* Wih1 = (const float*)d_in[5];
  const float* Whh1 = (const float*)d_in[6];
  const float* bih1 = (const float*)d_in[7];
  const float* bhh1 = (const float*)d_in[8];
  const float* Wih2 = (const float*)d_in[9];
  const float* Whh2 = (const float*)d_in[10];
  const float* bih2 = (const float*)d_in[11];
  const float* bhh2 = (const float*)d_in[12];
  const float* Wout = (const float*)d_in[13];
  const float* bout = (const float*)d_in[14];
  float* out = (float*)d_out;

  // workspace layout
  char* ws = (char*)d_ws;
  unsigned int*   counter = (unsigned int*)ws;                         // @0
  unsigned short* woutbf  = (unsigned short*)(ws + 4096);              // 121*1024 bf16
  const size_t hbytes = (size_t)B_ * T_ * H_ * sizeof(unsigned short); // 64 MB
  unsigned short* hA = (unsigned short*)(ws + (512u << 10));
  unsigned short* hB = (unsigned short*)(ws + (512u << 10) + hbytes);

  // 1) reset grid-barrier counter (ws is poisoned / stale between calls)
  zero_u32<<<1, 64, 0, stream>>>(counter);

  // 2) bf16 copy of W_out for the WMMA output projection
  {
    int n = OUT_ * H_;
    cvt_f32_bf16<<<(n + 255) / 256, 256, 0, stream>>>(Wout, woutbf, n);
  }

  // 3) persistent 3-layer LSTM
  {
    size_t smem = (size_t)(2 * 32 * KPAD) * sizeof(unsigned short)     // weights
                + (size_t)(2 * 64 * 33 + 512 + 32) * sizeof(float);    // gates/c/bias
    (void)hipFuncSetAttribute((const void*)lstm3_persistent,
                              hipFuncAttributeMaxDynamicSharedMemorySize,
                              (int)smem);
    lstm3_persistent<<<NWG, TBLK, smem, stream>>>(
        strokes, Wih0, Whh0, bih0, bhh0, Wih1, Whh1, bih1, bhh1,
        Wih2, Whh2, bih2, bhh2, hA, hB, counter);
  }

  // 4) output projection (reads final-layer history in hA)
  {
    size_t smem = (size_t)(128 * KPAD) * sizeof(unsigned short);
    (void)hipFuncSetAttribute((const void*)out_proj,
                              hipFuncAttributeMaxDynamicSharedMemorySize,
                              (int)smem);
    out_proj<<<(B_ * T_ / 16) / 8, 256, smem, stream>>>(hA, woutbf, bout, out);
  }

  (void)in_sizes; (void)n_in; (void)out_size; (void)ws_size;
}
